// AE_37117107372093
// MI455X (gfx1250) — matrix operand order repaired and checked by reading, not compiled, not run on hardware
//
#include <hip/hip_runtime.h>
#include <hip/hip_bf16.h>

typedef _Float16 v16h __attribute__((ext_vector_type(16)));
typedef float    v8f  __attribute__((ext_vector_type(8)));

#define NFEAT   16
#define PDIM    5
#define INCH    21
#define MPAIRS  131072
#define LSTEPS  16
#define NTOT    4194304

// ---------------------------------------------------------------------------
// Initial Feature -> output copy (F lives in d_out and is updated in place)
// ---------------------------------------------------------------------------
__global__ __launch_bounds__(256) void AE_copy_feat(const float4* __restrict__ src,
                                                    float4* __restrict__ dst) {
  size_t i = (size_t)blockIdx.x * blockDim.x + threadIdx.x;
  dst[i] = src[i];
}

// Sinusoid positional table entry: pos[step][c]
__device__ __forceinline__ float pos_val(int step, int c) {
  float expo  = (float)(c & ~1) * (1.0f / 16.0f);
  float denom = __powf(10000.0f, expo);
  float ang   = (float)step / denom;
  return (c & 1) ? __cosf(ang) : __sinf(ang);
}

// ---------------------------------------------------------------------------
// Fused gather + two 3-layer MLPs (left/right) + add, per 16-row tile / wave.
// WMMA D = A x B + C with A = activations (16 x K<=32), B = W^T fragment.
// ---------------------------------------------------------------------------
__global__ __launch_bounds__(256) void AE_fused_mlp(
    const float* __restrict__ X, const float* __restrict__ F,
    const long long* __restrict__ I,       // step-local, stride 3 per row
    const float* __restrict__ w1, const float* __restrict__ b1,
    const float* __restrict__ w2, const float* __restrict__ b2,
    const float* __restrict__ w3, const float* __restrict__ b3,
    float* __restrict__ tmp, int step)
{
  __shared__ float smem[8][16 * 17];       // per-wave 16x16 transpose tile (padded)
  const int lane = threadIdx.x & 31;
  const int wid  = threadIdx.x >> 5;
  const int n    = lane & 15;              // column (B/C/D) == row (A) for this lane
  const int hi   = lane >> 4;
  float* sm = smem[wid];

  // ---- Weight B-fragments (built once, live in VGPRs for whole kernel) ----
  // B layout: lane n = l&15 ; element e<8 maps to K = 8*hi + e, e>=8 maps to K = 16 + 8*hi + (e-8)
  v16h bw1, bw2, bw3;
#pragma unroll
  for (int e = 0; e < 16; ++e) {
    int k  = (e < 8) ? (hi * 8 + e) : (16 + hi * 8 + (e - 8));
    bw1[e] = (k < INCH)  ? (_Float16)w1[n * INCH  + k] : (_Float16)0.0f;
    bw2[e] = (k < NFEAT) ? (_Float16)w2[n * NFEAT + k] : (_Float16)0.0f;
    bw3[e] = (k < NFEAT) ? (_Float16)w3[n * NFEAT + k] : (_Float16)0.0f;
  }
  const float bias1 = b1[n], bias2 = b2[n], bias3 = b3[n];

  // ---- Layer-1 A-fragment K map (16-bit A 16x32 layout) ----
  // hi=0: e<8 -> K=e,    e>=8 -> K=16+(e-8)
  // hi=1: e<8 -> K=8+e,  e>=8 -> K=24+(e-8)
  const int kb0 = hi ? 8 : 0;
  const int kb1 = hi ? 24 : 16;
  int   kmap[16];
  float padd[16];
#pragma unroll
  for (int e = 0; e < 16; ++e) {
    int k   = (e < 8) ? (kb0 + e) : (kb1 + (e - 8));
    kmap[e] = k;
    padd[e] = (k >= PDIM && k < INCH) ? pos_val(step, k - PDIM) : 0.0f;
  }

  const int tile = blockIdx.x * 8 + wid;
  const int row  = tile * 16 + n;                    // batch row for this lane's A slot
  const long long li = I[(long long)row * 3 + 0];
  const long long ri = I[(long long)row * 3 + 1];

  auto run_side = [&](long long idx) -> v8f {
    // ---- layer 1: relu(in @ w1^T + b1), K=21 padded to 32 ----
    v16h a1;
#pragma unroll
    for (int e = 0; e < 16; ++e) {
      int   k = kmap[e];
      float x = 0.0f;
      if (k < PDIM)      x = X[idx * PDIM + k];
      else if (k < INCH) x = F[idx * NFEAT + (k - PDIM)] + padd[e];
      a1[e] = (_Float16)x;
    }
    v8f acc;
#pragma unroll
    for (int r = 0; r < 8; ++r) acc[r] = bias1;
    acc = __builtin_amdgcn_wmma_f32_16x16x32_f16(false, a1, false, bw1,
                                                 (short)0, acc, false, false);
    // D -> A transpose via LDS, ReLU fused on reload
#pragma unroll
    for (int r = 0; r < 8; ++r) sm[(r + 8 * hi) * 17 + n] = acc[r];
    asm volatile("s_wait_dscnt 0" ::: "memory");
    v16h a2;
#pragma unroll
    for (int e = 0; e < 16; ++e) {
      float x = (e < 8) ? fmaxf(sm[n * 17 + (hi * 8 + e)], 0.0f) : 0.0f;
      a2[e] = (_Float16)x;
    }
    // ---- layer 2: relu(h1 @ w2^T + b2), K=16 padded to 32 ----
#pragma unroll
    for (int r = 0; r < 8; ++r) acc[r] = bias2;
    acc = __builtin_amdgcn_wmma_f32_16x16x32_f16(false, a2, false, bw2,
                                                 (short)0, acc, false, false);
#pragma unroll
    for (int r = 0; r < 8; ++r) sm[(r + 8 * hi) * 17 + n] = acc[r];
    asm volatile("s_wait_dscnt 0" ::: "memory");
    v16h a3;
#pragma unroll
    for (int e = 0; e < 16; ++e) {
      float x = (e < 8) ? fmaxf(sm[n * 17 + (hi * 8 + e)], 0.0f) : 0.0f;
      a3[e] = (_Float16)x;
    }
    // ---- layer 3: h2 @ w3^T + b3 (no relu) ----
#pragma unroll
    for (int r = 0; r < 8; ++r) acc[r] = bias3;
    acc = __builtin_amdgcn_wmma_f32_16x16x32_f16(false, a3, false, bw3,
                                                 (short)0, acc, false, false);
    return acc;
  };

  v8f outL = run_side(li);
  v8f outR = run_side(ri);

#pragma unroll
  for (int r = 0; r < 8; ++r) {
    int m = r + 8 * hi;
    tmp[((size_t)tile * 16 + m) * NFEAT + n] = outL[r] + outR[r];
  }
}

// ---------------------------------------------------------------------------
// Scatter: F[pi[row]] = tmp[row]   (reads coalesced, 64B scattered writes)
// ---------------------------------------------------------------------------
__global__ __launch_bounds__(256) void AE_scatter(const float* __restrict__ tmp,
                                                  const long long* __restrict__ I,
                                                  float* __restrict__ F) {
  int j   = blockIdx.x * 256 + threadIdx.x;
  int row = j >> 4;
  int c   = j & 15;
  long long pi = I[(long long)row * 3 + 2];
  F[pi * NFEAT + c] = tmp[j];
}

extern "C" void kernel_launch(void* const* d_in, const int* in_sizes, int n_in,
                              void* d_out, int out_size, void* d_ws, size_t ws_size,
                              hipStream_t stream) {
  const float*     X  = (const float*)d_in[0];
  const float*     Fe = (const float*)d_in[1];
  const long long* I  = (const long long*)d_in[2];
  const float* w1 = (const float*)d_in[3];
  const float* b1 = (const float*)d_in[4];
  const float* w2 = (const float*)d_in[5];
  const float* b2 = (const float*)d_in[6];
  const float* w3 = (const float*)d_in[7];
  const float* b3 = (const float*)d_in[8];
  float* F   = (float*)d_out;        // running feature buffer == output
  float* tmp = (float*)d_ws;         // MPAIRS*16 floats = 8 MB scratch

  // F = Feature (full init every call -> deterministic across graph replays)
  size_t total4 = (size_t)NTOT * NFEAT / 4;
  AE_copy_feat<<<(unsigned)(total4 / 256), 256, 0, stream>>>(
      (const float4*)Fe, (float4*)F);

  for (int step = 0; step < LSTEPS; ++step) {
    const long long* Istep = I + (size_t)step * MPAIRS * 3;
    AE_fused_mlp<<<MPAIRS / 16 / 8, 256, 0, stream>>>(
        X, F, Istep, w1, b1, w2, b2, w3, b3, tmp, step);
    AE_scatter<<<MPAIRS * 16 / 256, 256, 0, stream>>>(tmp, Istep, F);
  }
}
